// Model3_1228360646820
// MI455X (gfx1250) — compile-verified, matrix operand order checked by reading, not run on hardware
//
#include <hip/hip_runtime.h>
#include <cstdint>
#include <cstddef>

// ---------------- problem dims ----------------
constexpr int kB = 32, kT = 64, kP = 10, kI = 1024, kH = 1024, kC = 8;
constexpr int kG = 4 * kH; // 4096 gate width

// ---------------- vector types ----------------
typedef __attribute__((ext_vector_type(16))) __bf16 v16bf;
typedef __attribute__((ext_vector_type(8)))  __bf16 v8bf;
typedef __attribute__((ext_vector_type(8)))  float  v8f;
typedef __attribute__((ext_vector_type(4)))  unsigned int u32x4;
typedef __attribute__((ext_vector_type(8)))  int i32x8;
typedef __attribute__((ext_vector_type(4)))  int i32x4;

// TDM availability: device pass only (host pass sees 0, bodies diverge safely)
#if __has_builtin(__builtin_amdgcn_tensor_load_to_lds)
#define ATHENA_TDM 1
#else
#define ATHENA_TDM 0
#endif

// fp32 -> bf16 round-to-nearest-even
__device__ inline __bf16 f2bf(float f) {
  uint32_t u = __builtin_bit_cast(uint32_t, f);
  u += 0x7FFFu + ((u >> 16) & 1u);
  uint16_t h16 = (uint16_t)(u >> 16);
  __bf16 r;
  __builtin_memcpy(&r, &h16, 2);
  return r;
}

__device__ inline float sigm(float x) { return 1.0f / (1.0f + expf(-x)); }

// Lane fragment: two 16-byte contiguous loads (matches CDNA5 16-bit A/B
// layout: VGPR0-3 hold K=half*8+0..7 contiguous, VGPR4-7 hold K=16+half*8+0..7).
__device__ __forceinline__ v16bf ld_frag16(const __bf16* lo, const __bf16* hi) {
  v8bf a = *(const v8bf*)lo;
  v8bf b = *(const v8bf*)hi;
  v16bf f;
#pragma unroll
  for (int i = 0; i < 8; ++i) { f[i] = a[i]; f[i + 8] = b[i]; }
  return f;
}

#if ATHENA_TDM
__device__ __forceinline__ void tdm_issue(u32x4 g0, i32x8 g1) {
  i32x4 z4 = {};
#if __clang_major__ >= 23
  i32x8 z8 = {};
  __builtin_amdgcn_tensor_load_to_lds(g0, g1, z4, z4, z8, 0);
#else
  __builtin_amdgcn_tensor_load_to_lds(g0, g1, z4, z4, 0);
#endif
}

// Stage a 16-row x 32-element bf16 tile (row stride = ldb elements) into LDS.
// D# per CDNA5 ISA 8.3/8.4: group0 = {count=1, lds_addr, 57b global addr,
// type=2}; group1 = {mask=0, data_size=2B, tile 32x16, dim0_stride}.
// Tensor dims set huge so no OOB clamping applies (tile fully in range).
__device__ __forceinline__ void tdm_load_b_tile(const __bf16* gsrc,
                                                uint32_t lds_off,
                                                int row_stride_elems) {
  const unsigned long long ga = (unsigned long long)(uintptr_t)gsrc;
  u32x4 g0;
  g0.x = 1u;                                        // count=1, is_restore=0
  g0.y = lds_off;                                   // LDS byte address
  g0.z = (unsigned)(ga & 0xFFFFFFFFull);            // global_addr[31:0]
  g0.w = (unsigned)((ga >> 32) & 0x01FFFFFFull)     // global_addr[56:32]
       | (2u << 30);                                // type=2 ("image")
  const unsigned td0 = 1u << 30, td1 = 1u << 30;    // huge tensor dims
  i32x8 g1;
  g1[0] = (int)(1u << 16);                          // data_size=1 (2 bytes)
  g1[1] = (int)((td0 & 0xFFFFu) << 16);             // tensor_dim0[15:0] @ [63:48]
  g1[2] = (int)(((td0 >> 16) & 0xFFFFu) | ((td1 & 0xFFFFu) << 16));
  g1[3] = (int)(((td1 >> 16) & 0xFFFFu) | (32u << 16)); // tile_dim0 = 32
  g1[4] = (int)16u;                                 // tile_dim1=16, tile_dim2=0
  g1[5] = (int)(unsigned)row_stride_elems;          // tensor_dim0_stride[31:0]
  g1[6] = 0;                                        // stride hi / dim1_stride
  g1[7] = 0;
  tdm_issue(g0, g1);
}
#endif // ATHENA_TDM

// ---------------------------------------------------------------------------
// C = A * B^T (+ A2 * B2^T) (+ bias[n]); bf16 inputs, fp32 accum.
// A-row mapping supports slicing x[:, t, :, :]:
//   rowOffset(m) = (m/aRowsPerGroup)*aGroupStride + (m%aRowsPerGroup)*aRowStride
// 8 waves/block, one 16x16 WMMA tile per wave, K stepped by 32.
// B1 panel is staged into LDS by the Tensor Data Mover with a per-wave double
// buffer. Steady-state loop is branch-free (next TDM issued unconditionally,
// wait tensorcnt<=1); the final K-chunk is a peeled epilogue (tensorcnt<=0).
// A fragments are software-pipelined one K-chunk ahead.
// ---------------------------------------------------------------------------
template <bool FUSED>
__global__ __launch_bounds__(256) void wmma_gemm_bt(
    const __bf16* __restrict__ A, long long aGroupStride, int aRowsPerGroup, int aRowStride,
    const __bf16* __restrict__ Bw, int ldb,
    const __bf16* __restrict__ A2, int a2RowStride,
    const __bf16* __restrict__ B2, int ldb2,
    const float* __restrict__ bias,
    float* __restrict__ C, int ldc,
    int M, int N, int K)
{
#if ATHENA_TDM
  __shared__ __align__(16) __bf16 ldsB[8][2][16 * 32]; // 16 KB: 8 waves x 2 bufs
#endif
  const int lane = threadIdx.x & 31;   // wave32
  const int wave = threadIdx.x >> 5;
  const int tilesN = N >> 4;
  const int tile = blockIdx.x * 8 + wave;
  const int tM = tile / tilesN;
  const int tN = tile % tilesN;
  if (tM * 16 >= M) return;            // whole-wave exit: EXEC stays all-ones

  const int row = lane & 15;
  const int hf  = lane >> 4;

  const int mr = tM * 16 + row;
  const __bf16* arow = A + (long long)(mr / aRowsPerGroup) * aGroupStride
                         + (long long)(mr % aRowsPerGroup) * aRowStride;
  const __bf16* a2row = FUSED ? (A2 + (long long)mr * a2RowStride) : (const __bf16*)nullptr;
  const __bf16* b2row = FUSED ? (B2 + (long long)(tN * 16 + row) * ldb2) : (const __bf16*)nullptr;

  v8f acc = {};

#if ATHENA_TDM
  const __bf16* btile = Bw + (long long)(tN * 16) * ldb;     // 16 x K panel
  __bf16* lbuf0 = &ldsB[wave][0][0];
  __bf16* lbuf1 = &ldsB[wave][1][0];
  // generic LDS address: low 32 bits == byte offset in LDS aperture
  const uint32_t loff0 = (uint32_t)(uintptr_t)lbuf0;
  const uint32_t loff1 = (uint32_t)(uintptr_t)lbuf1;

  tdm_load_b_tile(btile, loff0, ldb);                        // prime buffer 0
  v16bf fa = ld_frag16(arow + hf * 8, arow + 16 + hf * 8);   // prime A frag

  int buf = 0;
  int k0 = 0;
  // ---- branch-free steady state: next chunk always exists ----
  for (; k0 + 32 < K; k0 += 32) {
    tdm_load_b_tile(btile + k0 + 32, buf ? loff0 : loff1, ldb); // DMA ahead
    __builtin_prefetch(arow + k0 + 288, 0, 0);                  // streaming A
    v16bf fa_next = ld_frag16(arow + k0 + 32 + hf * 8,
                              arow + k0 + 48 + hf * 8);

    __builtin_amdgcn_s_wait_tensorcnt(1);   // current buf done; next in flight
    const __bf16* lb = buf ? lbuf1 : lbuf0;
    v16bf fb = ld_frag16(lb + row * 32 + hf * 8, lb + row * 32 + 16 + hf * 8);
    acc = __builtin_amdgcn_wmma_f32_16x16x32_bf16(
        false, fa, false, fb, (short)0, acc, false, false);

    if (FUSED) {
      v16bf fa2 = ld_frag16(a2row + k0 + hf * 8, a2row + k0 + 16 + hf * 8);
      v16bf fb2 = ld_frag16(b2row + k0 + hf * 8, b2row + k0 + 16 + hf * 8);
      acc = __builtin_amdgcn_wmma_f32_16x16x32_bf16(
          false, fa2, false, fb2, (short)0, acc, false, false);
    }
    fa = fa_next;
    buf ^= 1;
  }
  // ---- peeled final chunk ----
  {
    __builtin_amdgcn_s_wait_tensorcnt(0);
    const __bf16* lb = buf ? lbuf1 : lbuf0;
    v16bf fb = ld_frag16(lb + row * 32 + hf * 8, lb + row * 32 + 16 + hf * 8);
    acc = __builtin_amdgcn_wmma_f32_16x16x32_bf16(
        false, fa, false, fb, (short)0, acc, false, false);
    if (FUSED) {
      v16bf fa2 = ld_frag16(a2row + k0 + hf * 8, a2row + k0 + 16 + hf * 8);
      v16bf fb2 = ld_frag16(b2row + k0 + hf * 8, b2row + k0 + 16 + hf * 8);
      acc = __builtin_amdgcn_wmma_f32_16x16x32_bf16(
          false, fa2, false, fb2, (short)0, acc, false, false);
    }
  }
#else
  const __bf16* brow = Bw + (long long)(tN * 16 + row) * ldb;
  for (int k0 = 0; k0 < K; k0 += 32) {
    __builtin_prefetch(arow + k0 + 256, 0, 0);
    v16bf fa = ld_frag16(arow + k0 + hf * 8, arow + k0 + 16 + hf * 8);
    v16bf fb = ld_frag16(brow + k0 + hf * 8, brow + k0 + 16 + hf * 8);
    acc = __builtin_amdgcn_wmma_f32_16x16x32_bf16(
        false, fa, false, fb, (short)0, acc, false, false);
    if (FUSED) {
      v16bf fa2 = ld_frag16(a2row + k0 + hf * 8, a2row + k0 + 16 + hf * 8);
      v16bf fb2 = ld_frag16(b2row + k0 + hf * 8, b2row + k0 + 16 + hf * 8);
      acc = __builtin_amdgcn_wmma_f32_16x16x32_bf16(
          false, fa2, false, fb2, (short)0, acc, false, false);
    }
  }
#endif

  // C/D layout: lane covers col n = tN*16 + (lane&15),
  // rows m = tM*16 + (lane>>4)*8 + r for accumulator VGPR r.
  const int ncol = tN * 16 + row;
  const float bv = bias ? bias[ncol] : 0.0f;
#pragma unroll
  for (int r = 0; r < 8; ++r) {
    const int mrow = tM * 16 + hf * 8 + r;
    C[(long long)mrow * ldc + ncol] = acc[r] + bv;
  }
}

// ---------------------------------------------------------------------------
// Attention: scores_p = v . tanh(E[b,p,:] + HW[b,:])  (b_att folded into E),
// softmax over P, emb[b,:] = sum_p alpha_p * x[b,t,p,:]  -> bf16 for next GEMM
// ---------------------------------------------------------------------------
__global__ __launch_bounds__(256) void attention_kernel(
    const float* __restrict__ E,    // (B*P, H), already + b_att
    const float* __restrict__ HW,   // (B, H)
    const float* __restrict__ v,    // (H)
    const float* __restrict__ x,    // (B, T, P, I) fp32
    int t,
    __bf16* __restrict__ emb_bf)    // (B, I)
{
  const int b = blockIdx.x;
  const int tid = threadIdx.x;
  __shared__ float red[256];
  __shared__ float alpha[kP];

  float accp[kP];
#pragma unroll
  for (int p = 0; p < kP; ++p) accp[p] = 0.0f;

  for (int h = tid; h < kH; h += 256) {
    const float base = HW[(size_t)b * kH + h];
    const float vh = v[h];
#pragma unroll
    for (int p = 0; p < kP; ++p)
      accp[p] += vh * tanhf(E[((size_t)(b * kP + p)) * kH + h] + base);
  }
  for (int p = 0; p < kP; ++p) {
    red[tid] = accp[p];
    __syncthreads();
    for (int s = 128; s > 0; s >>= 1) {
      if (tid < s) red[tid] += red[tid + s];
      __syncthreads();
    }
    if (tid == 0) alpha[p] = red[0];
    __syncthreads();
  }
  if (tid == 0) {
    float mx = alpha[0];
    for (int p = 1; p < kP; ++p) mx = fmaxf(mx, alpha[p]);
    float s = 0.0f;
    for (int p = 0; p < kP; ++p) { alpha[p] = expf(alpha[p] - mx); s += alpha[p]; }
    const float inv = 1.0f / s;
    for (int p = 0; p < kP; ++p) alpha[p] *= inv;
  }
  __syncthreads();

  const float* xb = x + (((size_t)b * kT + t) * kP) * kI;
  for (int i = tid; i < kI; i += 256) {
    float e = 0.0f;
#pragma unroll
    for (int p = 0; p < kP; ++p) e += alpha[p] * xb[(size_t)p * kI + i];
    emb_bf[(size_t)b * kI + i] = f2bf(e);
  }
}

// LSTM pointwise: PyTorch gate order i,f,g,o
__global__ __launch_bounds__(256) void lstm_kernel(
    const float* __restrict__ G,  // (B, 4H)
    float* __restrict__ c,        // (B, H)
    float* __restrict__ h,        // (B, H)
    __bf16* __restrict__ h_bf)    // (B, H)
{
  const int idx = blockIdx.x * 256 + threadIdx.x;
  if (idx >= kB * kH) return;
  const int b = idx / kH, hi = idx % kH;
  const float* g = G + (size_t)b * kG;
  const float i_ = g[hi];
  const float f_ = g[kH + hi];
  const float gg = g[2 * kH + hi];
  const float o_ = g[3 * kH + hi];
  const float cn = sigm(f_) * c[idx] + sigm(i_) * tanhf(gg);
  const float hn = sigm(o_) * tanhf(cn);
  c[idx] = cn;
  h[idx] = hn;
  h_bf[idx] = f2bf(hn);
}

// Final FC: (B,H) @ (C,H)^T + bfc -> (B,C) = 256 outputs, one thread each.
__global__ void fc_kernel(const float* __restrict__ h,
                          const float* __restrict__ Wfc,
                          const float* __restrict__ bfc,
                          float* __restrict__ out)
{
  const int idx = threadIdx.x;         // 0..B*C-1
  const int b = idx / kC, cc = idx % kC;
  float a = bfc[cc];
  for (int k = 0; k < kH; ++k)
    a += h[(size_t)b * kH + k] * Wfc[(size_t)cc * kH + k];
  out[idx] = a;
}

__global__ __launch_bounds__(256) void cvt_bf16(const float* __restrict__ s,
                                                __bf16* __restrict__ d,
                                                long long n)
{
  long long i = (long long)blockIdx.x * 256 + threadIdx.x;
  if (i < n) d[i] = f2bf(s[i]);
}

__global__ void add_bias(const float* __restrict__ a, const float* __restrict__ b,
                         float* __restrict__ o, int n)
{
  int i = blockIdx.x * 256 + threadIdx.x;
  if (i < n) o[i] = a[i] + b[i];
}

// ---------------------------------------------------------------------------
extern "C" void kernel_launch(void* const* d_in, const int* in_sizes, int n_in,
                              void* d_out, int out_size, void* d_ws, size_t ws_size,
                              hipStream_t stream)
{
  (void)in_sizes; (void)n_in; (void)out_size; (void)ws_size;
  const float* x     = (const float*)d_in[0];
  const float* Wx    = (const float*)d_in[1];
  const float* Wh    = (const float*)d_in[2];
  const float* b_att = (const float*)d_in[3];
  const float* v     = (const float*)d_in[4];
  const float* Wih   = (const float*)d_in[5];
  const float* Whh   = (const float*)d_in[6];
  const float* bih   = (const float*)d_in[7];
  const float* bhh   = (const float*)d_in[8];
  const float* Wfc   = (const float*)d_in[9];
  const float* bfc   = (const float*)d_in[10];

  // ---- workspace carve-out (256B aligned) ----
  uintptr_t w = (uintptr_t)d_ws;
  auto take = [&](size_t bytes) -> void* {
    uintptr_t p = (w + 255) & ~(uintptr_t)255;
    w = p + bytes;
    return (void*)p;
  };
  const size_t NX = (size_t)kB * kT * kP * kI;
  __bf16* x_bf   = (__bf16*)take(NX * 2);
  __bf16* Wx_bf  = (__bf16*)take((size_t)kH * kI * 2);
  __bf16* Wh_bf  = (__bf16*)take((size_t)kH * kH * 2);
  __bf16* Wih_bf = (__bf16*)take((size_t)kG * kI * 2);
  __bf16* Whh_bf = (__bf16*)take((size_t)kG * kH * 2);
  float*  E      = (float*)take((size_t)kB * kP * kH * 4);
  float*  HWm    = (float*)take((size_t)kB * kH * 4);
  float*  Gm     = (float*)take((size_t)kB * kG * 4);
  float*  hS     = (float*)take((size_t)kB * kH * 4);
  float*  cS     = (float*)take((size_t)kB * kH * 4);
  __bf16* h_bf   = (__bf16*)take((size_t)kB * kH * 2);
  __bf16* emb_bf = (__bf16*)take((size_t)kB * kH * 2);
  float*  biasG  = (float*)take((size_t)kG * 4);

  auto cvt = [&](const float* s, __bf16* d, size_t n) {
    cvt_bf16<<<(int)((n + 255) / 256), 256, 0, stream>>>(s, d, (long long)n);
  };

  // One-time conversions (weights stay resident in the 192MB L2 across steps)
  cvt(x,   x_bf,   NX);
  cvt(Wx,  Wx_bf,  (size_t)kH * kI);
  cvt(Wh,  Wh_bf,  (size_t)kH * kH);
  cvt(Wih, Wih_bf, (size_t)kG * kI);
  cvt(Whh, Whh_bf, (size_t)kG * kH);
  add_bias<<<(kG + 255) / 256, 256, 0, stream>>>(bih, bhh, biasG, kG);

  hipMemsetAsync(hS,   0, (size_t)kB * kH * 4, stream);
  hipMemsetAsync(cS,   0, (size_t)kB * kH * 4, stream);
  hipMemsetAsync(h_bf, 0, (size_t)kB * kH * 2, stream);

  const int kBIG = 0x40000000; // "no grouping" for contiguous A matrices

  for (int t = 0; t < kT; ++t) {
    // E = x_t @ Wx^T + b_att   (M = B*P = 320, N = H, K = I)
    {
      const int tiles = ((kB * kP) / 16) * (kH / 16);
      wmma_gemm_bt<false><<<(tiles + 7) / 8, 256, 0, stream>>>(
          x_bf + (size_t)t * kP * kI, (long long)kT * kP * kI, kP, kI,
          Wx_bf, kI, nullptr, 0, nullptr, 0,
          b_att, E, kH, kB * kP, kH, kI);
    }
    // HW = h @ Wh^T            (M = B = 32, N = H, K = H)
    {
      const int tiles = (kB / 16) * (kH / 16);
      wmma_gemm_bt<false><<<(tiles + 7) / 8, 256, 0, stream>>>(
          h_bf, 0, kBIG, kH, Wh_bf, kH, nullptr, 0, nullptr, 0,
          nullptr, HWm, kH, kB, kH, kH);
    }

    attention_kernel<<<kB, 256, 0, stream>>>(E, HWm, v, x, t, emb_bf);

    // G = emb @ Wih^T + h @ Whh^T + (bih+bhh)   (M = 32, N = 4H, K = 1024)
    {
      const int tiles = (kB / 16) * (kG / 16);
      wmma_gemm_bt<true><<<(tiles + 7) / 8, 256, 0, stream>>>(
          emb_bf, 0, kBIG, kI, Wih_bf, kI,
          h_bf, kH, Whh_bf, kH,
          biasG, Gm, kG, kB, kG, kI);
    }

    lstm_kernel<<<(kB * kH + 255) / 256, 256, 0, stream>>>(Gm, cS, hS, h_bf);
  }

  fc_kernel<<<1, kB * kC, 0, stream>>>(hS, Wfc, bfc, (float*)d_out);
}
